// Model_38087769981557
// MI455X (gfx1250) — matrix operand order repaired and checked by reading, not compiled, not run on hardware
//
#include <hip/hip_runtime.h>
#include <hip/hip_bf16.h>
#include <math.h>

typedef __attribute__((ext_vector_type(16))) __bf16 v16bf;
typedef __attribute__((ext_vector_type(8)))  float  v8f;

#define BN_EPS 1e-5f

// ---------- helpers ----------
__device__ __forceinline__ unsigned pack_bf16x2(float a, float b) {
  union { float f; unsigned u; } ua, ub;
  ua.f = a; ub.f = b;
  unsigned ha = (ua.u + 0x7FFFu + ((ua.u >> 16) & 1u)) >> 16;  // RNE f32->bf16
  unsigned hb = (ub.u + 0x7FFFu + ((ub.u >> 16) & 1u)) >> 16;
  return (ha & 0xFFFFu) | (hb << 16);
}

__device__ __forceinline__ float4 ldg4_guard(const float* row, bool rv, int k, int K) {
  float4 r = {0.f, 0.f, 0.f, 0.f};
  if (!rv) return r;
  if (k + 3 < K) {
    r = *(const float4*)(row + k);
  } else {
    if (k     < K) r.x = row[k];
    if (k + 1 < K) r.y = row[k + 1];
    if (k + 2 < K) r.z = row[k + 2];
  }
  return r;
}

__device__ __forceinline__ float act_softplus(float x) {
  return (x > 20.f) ? x : log1pf(expf(x));
}

// ---------- bf16 WMMA GEMM: C[MxN] = act(A[MxK] * op(B) + bias) ----------
// BT=true : B is [N x K] (weight [out,in]), op(B) = B^T
// BT=false: B is [K x N] row-major
template <bool BT, int ACT>
__global__ __launch_bounds__(256)
void gemm_wmma_bf16(const float* __restrict__ A, const float* __restrict__ B,
                    const float* __restrict__ bias, float* __restrict__ C,
                    int M, int N, int K) {
  constexpr int BM = 128, BN = 64, BK = 64;
  // uint (bf16-pair) row stride: 32 data + 4 pad = 36 -> rows 144B (16B aligned,
  // 36 mod 64 banks => 16 row-lanes hit 16 distinct banks)
  constexpr int LDU = BK / 2 + 4;
  __shared__ unsigned Asu[BM * LDU];
  __shared__ unsigned Bsu[BN * LDU];

  const int tid  = threadIdx.x;
  const int lane = tid & 31;
  const int wave = tid >> 5;
  const int wm   = (wave & 3) * 32;  // 4 wave-tiles along M
  const int wn   = (wave >> 2) * 32; // 2 wave-tiles along N
  const int bm0  = blockIdx.y * BM;
  const int bn0  = blockIdx.x * BN;
  const int h    = lane >> 4;
  const int l16  = lane & 15;

  v8f acc[2][2] = {};

  for (int kt = 0; kt < K; kt += BK) {
    // ---- stage A tile (BM x BK): f32 -> packed bf16 pairs, b64 LDS stores ----
#pragma unroll
    for (int it = 0; it < 8; ++it) {
      int idx = it * 256 + tid;
      int ml = idx >> 4;
      int k4 = (idx & 15) << 2;
      int m  = bm0 + ml;
      float4 v = ldg4_guard(A + (size_t)m * K, m < M, kt + k4, K);
      uint2 p; p.x = pack_bf16x2(v.x, v.y); p.y = pack_bf16x2(v.z, v.w);
      *(uint2*)&Asu[ml * LDU + (k4 >> 1)] = p;
    }
    // ---- stage B tile -> Bs[n][k] ----
    if (BT) {
#pragma unroll
      for (int it = 0; it < 4; ++it) {
        int idx = it * 256 + tid;
        int nl = idx >> 4;
        int k4 = (idx & 15) << 2;
        int n  = bn0 + nl;
        float4 v = ldg4_guard(B + (size_t)n * K, n < N, kt + k4, K);
        uint2 p; p.x = pack_bf16x2(v.x, v.y); p.y = pack_bf16x2(v.z, v.w);
        *(uint2*)&Bsu[nl * LDU + (k4 >> 1)] = p;
      }
    } else {
      // B row-major [K x N]: read 2 k-rows x 4 n, transpose-pack into Bs[n][k]
#pragma unroll
      for (int it = 0; it < 2; ++it) {
        int idx = it * 256 + tid;
        int kp = idx >> 4;           // k-pair 0..31
        int n4 = (idx & 15) << 2;
        int k  = kt + kp * 2;
        float4 v0 = ldg4_guard(B + (size_t)k * N,       k     < K, bn0 + n4, N);
        float4 v1 = ldg4_guard(B + (size_t)(k + 1) * N, (k+1) < K, bn0 + n4, N);
        Bsu[(n4 + 0) * LDU + kp] = pack_bf16x2(v0.x, v1.x);
        Bsu[(n4 + 1) * LDU + kp] = pack_bf16x2(v0.y, v1.y);
        Bsu[(n4 + 2) * LDU + kp] = pack_bf16x2(v0.z, v1.z);
        Bsu[(n4 + 3) * LDU + kp] = pack_bf16x2(v0.w, v1.w);
      }
    }
    __syncthreads();

    // ---- 2 WMMA K-steps per stage; fragments via ds_load_b128 ----
#pragma unroll
    for (int kk = 0; kk < BK; kk += 32) {
      union { v16bf v; uint4 q[2]; } afr[2], bfr[2];
      // A 16x32 bf16: u[v]: K = kk + g*16 + h*8 + q*2
      //   -> u[0..3] = row[kk/2 + h*4 .. +3], u[4..7] = row[kk/2 + 8 + h*4 .. +3]
#pragma unroll
      for (int i = 0; i < 2; ++i) {
        const unsigned* base = &Asu[(wm + i * 16 + l16) * LDU + (kk >> 1)];
        afr[i].q[0] = *(const uint4*)(base + h * 4);
        afr[i].q[1] = *(const uint4*)(base + 8 + h * 4);
      }
      // B 32x16 bf16: same K order as A -> u[0..3] = row[kk/2 + h*4 .. +3], u[4..7] = row[kk/2 + 8 + h*4 .. +3]
#pragma unroll
      for (int j = 0; j < 2; ++j) {
        const unsigned* base = &Bsu[(wn + j * 16 + l16) * LDU + (kk >> 1)];
        bfr[j].q[0] = *(const uint4*)(base + h * 4);
        bfr[j].q[1] = *(const uint4*)(base + 8 + h * 4);
      }
#pragma unroll
      for (int i = 0; i < 2; ++i)
#pragma unroll
        for (int j = 0; j < 2; ++j)
          acc[i][j] = __builtin_amdgcn_wmma_f32_16x16x32_bf16(
              false, afr[i].v, false, bfr[j].v, (short)0, acc[i][j], false, false);
    }
    __syncthreads();
  }

  // ---- epilogue: C/D layout: n = l16, m = 8*h + r ----
#pragma unroll
  for (int i = 0; i < 2; ++i) {
#pragma unroll
    for (int j = 0; j < 2; ++j) {
      int n = bn0 + wn + j * 16 + l16;
      if (n >= N) continue;
      float bv = bias ? bias[n] : 0.f;
      int mbase = bm0 + wm + i * 16 + h * 8;
#pragma unroll
      for (int r = 0; r < 8; ++r) {
        int m = mbase + r;
        if (m < M) {
          float v = acc[i][j][r] + bv;
          if (ACT == 1) v = act_softplus(v);
          C[(size_t)m * N + n] = v;
        }
      }
    }
  }
}

// ---------- mean/logvar linear heads (C=2, fused) ----------
__global__ __launch_bounds__(128)
void meanlogvar_kernel(const float* __restrict__ en2,
                       const float* __restrict__ mW, const float* __restrict__ mb,
                       const float* __restrict__ lW, const float* __restrict__ lb,
                       float* __restrict__ mean_lin, float* __restrict__ logvar_lin,
                       int E2) {
  __shared__ float red[128];
  int n = blockIdx.x, tid = threadIdx.x;
  const float* row = en2 + (size_t)n * E2;
  float a0 = 0, a1 = 0, a2 = 0, a3 = 0;
  for (int k = tid; k < E2; k += 128) {
    float e = row[k];
    a0 += e * mW[k]; a1 += e * mW[E2 + k];
    a2 += e * lW[k]; a3 += e * lW[E2 + k];
  }
  float acc[4] = {a0, a1, a2, a3}, res[4];
  for (int q = 0; q < 4; ++q) {
    red[tid] = acc[q]; __syncthreads();
    for (int s = 64; s > 0; s >>= 1) { if (tid < s) red[tid] += red[tid + s]; __syncthreads(); }
    res[q] = red[0]; __syncthreads();
  }
  if (tid == 0) {
    mean_lin[2 * n]     = res[0] + mb[0];
    mean_lin[2 * n + 1] = res[1] + mb[1];
    logvar_lin[2 * n]     = res[2] + lb[0];
    logvar_lin[2 * n + 1] = res[3] + lb[1];
  }
}

// ---------- fused latent path: BN(mean), BN(logvar), z, BN(z)->zx, BN(topics)->zc ----------
__global__ __launch_bounds__(1024)
void latent_kernel(const float* __restrict__ mean_lin, const float* __restrict__ logvar_lin,
                   const float* __restrict__ eps_in, const float* __restrict__ topics,
                   const float* __restrict__ g_mean, const float* __restrict__ b_mean,
                   const float* __restrict__ g_logvar, const float* __restrict__ b_logvar,
                   const float* __restrict__ g_x, const float* __restrict__ b_x,
                   const float* __restrict__ g_phi, const float* __restrict__ b_phi,
                   float* __restrict__ z_out, float* __restrict__ zx_out,
                   float* __restrict__ zc_out, int N, int Kt) {
  __shared__ float red[1024];
  __shared__ float st[8];
  int tid = threadIdx.x;

  auto block_reduce = [&](const float* vals, int cnt) {
    for (int q = 0; q < cnt; ++q) {
      red[tid] = vals[q]; __syncthreads();
      for (int s = 512; s > 0; s >>= 1) { if (tid < s) red[tid] += red[tid + s]; __syncthreads(); }
      if (tid == 0) st[q] = red[0];
      __syncthreads();
    }
  };

  // phase 1: batch stats of mean_lin / logvar_lin (2 cols each)
  float a[8] = {0, 0, 0, 0, 0, 0, 0, 0};
  for (int i = tid; i < N; i += 1024) {
    float m0 = mean_lin[2 * i], m1 = mean_lin[2 * i + 1];
    float l0 = logvar_lin[2 * i], l1 = logvar_lin[2 * i + 1];
    a[0] += m0; a[1] += m1; a[2] += m0 * m0; a[3] += m1 * m1;
    a[4] += l0; a[5] += l1; a[6] += l0 * l0; a[7] += l1 * l1;
  }
  block_reduce(a, 8);
  float invN = 1.f / (float)N;
  float mm0 = st[0] * invN, mm1 = st[1] * invN;
  float vm0 = st[2] * invN - mm0 * mm0, vm1 = st[3] * invN - mm1 * mm1;
  float ml0 = st[4] * invN, ml1 = st[5] * invN;
  float vl0 = st[6] * invN - ml0 * ml0, vl1 = st[7] * invN - ml1 * ml1;
  float scm0 = g_mean[0] * rsqrtf(vm0 + BN_EPS), scm1 = g_mean[1] * rsqrtf(vm1 + BN_EPS);
  float shm0 = b_mean[0] - mm0 * scm0, shm1 = b_mean[1] - mm1 * scm1;
  float scl0 = g_logvar[0] * rsqrtf(vl0 + BN_EPS), scl1 = g_logvar[1] * rsqrtf(vl1 + BN_EPS);
  float shl0 = b_logvar[0] - ml0 * scl0, shl1 = b_logvar[1] - ml1 * scl1;

  // phase 2: z = bn(mean) + sqrt(exp(bn(logvar))) * eps; accumulate z stats
  float zs[4] = {0, 0, 0, 0};
  for (int i = tid; i < N; i += 1024) {
    float pm0 = mean_lin[2 * i] * scm0 + shm0;
    float pm1 = mean_lin[2 * i + 1] * scm1 + shm1;
    float pl0 = logvar_lin[2 * i] * scl0 + shl0;
    float pl1 = logvar_lin[2 * i + 1] * scl1 + shl1;
    float z0 = pm0 + sqrtf(expf(pl0)) * eps_in[2 * i];
    float z1 = pm1 + sqrtf(expf(pl1)) * eps_in[2 * i + 1];
    z_out[2 * i] = z0; z_out[2 * i + 1] = z1;
    zs[0] += z0; zs[1] += z1; zs[2] += z0 * z0; zs[3] += z1 * z1;
  }
  block_reduce(zs, 4);
  float mz0 = st[0] * invN, mz1 = st[1] * invN;
  float vz0 = st[2] * invN - mz0 * mz0, vz1 = st[3] * invN - mz1 * mz1;
  float scx0 = g_x[0] * rsqrtf(vz0 + BN_EPS), scx1 = g_x[1] * rsqrtf(vz1 + BN_EPS);
  float shx0 = b_x[0] - mz0 * scx0, shx1 = b_x[1] - mz1 * scx1;
  for (int i = tid; i < N; i += 1024) {  // same-thread reads of own z writes
    zx_out[2 * i]     = z_out[2 * i]     * scx0 + shx0;
    zx_out[2 * i + 1] = z_out[2 * i + 1] * scx1 + shx1;
  }

  // phase 3: zc = BN(topics) over K rows
  float ts[4] = {0, 0, 0, 0};
  for (int i = tid; i < Kt; i += 1024) {
    float t0 = topics[2 * i], t1 = topics[2 * i + 1];
    ts[0] += t0; ts[1] += t1; ts[2] += t0 * t0; ts[3] += t1 * t1;
  }
  block_reduce(ts, 4);
  float invK = 1.f / (float)Kt;
  float mt0 = st[0] * invK, mt1 = st[1] * invK;
  float vt0 = st[2] * invK - mt0 * mt0, vt1 = st[3] * invK - mt1 * mt1;
  float scp0 = g_phi[0] * rsqrtf(vt0 + BN_EPS), scp1 = g_phi[1] * rsqrtf(vt1 + BN_EPS);
  float shp0 = b_phi[0] - mt0 * scp0, shp1 = b_phi[1] - mt1 * scp1;
  for (int i = tid; i < Kt; i += 1024) {
    zc_out[2 * i]     = topics[2 * i]     * scp0 + shp0;
    zc_out[2 * i + 1] = topics[2 * i + 1] * scp1 + shp1;
  }
}

// ---------- theta: RBF distances (C=2) + softmax over K ----------
__global__ __launch_bounds__(256)
void theta_kernel(const float* __restrict__ zx, const float* __restrict__ zc,
                  float* __restrict__ theta, int Kt) {
  __shared__ float zcs[512];
  __shared__ float red[256];
  int n = blockIdx.x, tid = threadIdx.x;
  for (int i = tid; i < 2 * Kt; i += 256) zcs[i] = zc[i];
  __syncthreads();
  float zx0 = zx[2 * n], zx1 = zx[2 * n + 1];
  float d = -3.4e38f;
  if (tid < Kt) {
    float dx = zx0 - zcs[2 * tid], dy = zx1 - zcs[2 * tid + 1];
    d = -0.5f * (dx * dx + dy * dy);
  }
  red[tid] = d; __syncthreads();
  for (int s = 128; s > 0; s >>= 1) { if (tid < s) red[tid] = fmaxf(red[tid], red[tid + s]); __syncthreads(); }
  float mx = red[0]; __syncthreads();
  float e = (tid < Kt) ? expf(d - mx) : 0.f;
  red[tid] = e; __syncthreads();
  for (int s = 128; s > 0; s >>= 1) { if (tid < s) red[tid] += red[tid + s]; __syncthreads(); }
  float sum = red[0];
  if (tid < Kt) theta[(size_t)n * Kt + tid] = e / sum;
}

// ---------- tiny GEMMs for mu chain (K <= 100) ----------
template <int ACT>
__global__ void small_gemm_bt(const float* __restrict__ A, const float* __restrict__ B,
                              const float* __restrict__ bias, float* __restrict__ C,
                              int M, int N, int K) {
  int idx = blockIdx.x * blockDim.x + threadIdx.x;
  if (idx >= M * N) return;
  int m = idx / N, n = idx % N;
  float s = bias ? bias[n] : 0.f;
  const float* ar = A + (size_t)m * K;
  const float* br = B + (size_t)n * K;
  for (int k = 0; k < K; ++k) s += ar[k] * br[k];
  if (ACT == 1) s = act_softplus(s);
  C[idx] = s;
}

// ---------- per-topic BN over V + softmax over V (logits stays L2-resident) ----------
__global__ __launch_bounds__(1024)
void beta_kernel(const float* __restrict__ logits, const float* __restrict__ bb,
                 const float* __restrict__ g_dec, const float* __restrict__ b_dec,
                 float* __restrict__ beta, int V) {
  __shared__ float red[1024];
  __shared__ float st[2];
  int k = blockIdx.x, tid = threadIdx.x;
  const float* row = logits + (size_t)k * V;
  const float* bbr = bb + (size_t)k * V;
  float s = 0, ss = 0;
  for (int v = tid; v < V; v += 1024) { float x = row[v]; s += x; ss += x * x; }
  red[tid] = s; __syncthreads();
  for (int t = 512; t > 0; t >>= 1) { if (tid < t) red[tid] += red[tid + t]; __syncthreads(); }
  if (tid == 0) st[0] = red[0]; __syncthreads();
  red[tid] = ss; __syncthreads();
  for (int t = 512; t > 0; t >>= 1) { if (tid < t) red[tid] += red[tid + t]; __syncthreads(); }
  if (tid == 0) st[1] = red[0]; __syncthreads();
  float mean = st[0] / V;
  float var  = st[1] / V - mean * mean;
  float scale = g_dec[k] * rsqrtf(var + BN_EPS);
  float shift = b_dec[k] - mean * scale;
  float mx = -3.4e38f;
  for (int v = tid; v < V; v += 1024) { float x = row[v] * scale + shift + bbr[v]; mx = fmaxf(mx, x); }
  red[tid] = mx; __syncthreads();
  for (int t = 512; t > 0; t >>= 1) { if (tid < t) red[tid] = fmaxf(red[tid], red[tid + t]); __syncthreads(); }
  mx = red[0]; __syncthreads();
  float es = 0;
  for (int v = tid; v < V; v += 1024) { float x = row[v] * scale + shift + bbr[v]; es += expf(x - mx); }
  red[tid] = es; __syncthreads();
  for (int t = 512; t > 0; t >>= 1) { if (tid < t) red[tid] += red[tid + t]; __syncthreads(); }
  float inv = 1.f / red[0];
  for (int v = tid; v < V; v += 1024) {
    float x = row[v] * scale + shift + bbr[v];
    beta[(size_t)k * V + v] = expf(x - mx) * inv;
  }
}

extern "C" void kernel_launch(void* const* d_in, const int* in_sizes, int n_in,
                              void* d_out, int out_size, void* d_ws, size_t ws_size,
                              hipStream_t stream) {
  const float* input_   = (const float*)d_in[0];
  // d_in[1] normalized_input_ is unused by the reference
  const float* eps      = (const float*)d_in[2];
  const float* en1_W    = (const float*)d_in[3];
  const float* en1_b    = (const float*)d_in[4];
  const float* en2_W    = (const float*)d_in[5];
  const float* en2_b    = (const float*)d_in[6];
  const float* mean_W   = (const float*)d_in[7];
  const float* mean_b   = (const float*)d_in[8];
  const float* logvar_W = (const float*)d_in[9];
  const float* logvar_b = (const float*)d_in[10];
  const float* mu1_W    = (const float*)d_in[11];
  const float* mu1_b    = (const float*)d_in[12];
  const float* mu2_W    = (const float*)d_in[13];
  const float* mu2_b    = (const float*)d_in[14];
  const float* mu_W     = (const float*)d_in[15];
  const float* mu_b     = (const float*)d_in[16];
  const float* topics   = (const float*)d_in[17];
  const float* beta_bias = (const float*)d_in[18];
  const float* emb      = (const float*)d_in[19];
  const float* g_mean   = (const float*)d_in[20];
  const float* b_mean   = (const float*)d_in[21];
  const float* g_logvar = (const float*)d_in[22];
  const float* b_logvar = (const float*)d_in[23];
  const float* g_x      = (const float*)d_in[24];
  const float* b_x      = (const float*)d_in[25];
  const float* g_phi    = (const float*)d_in[26];
  const float* b_phi    = (const float*)d_in[27];
  const float* g_dec    = (const float*)d_in[28];
  const float* b_dec    = (const float*)d_in[29];

  const int N = 2048, V = 50000, E1 = 1024, E2 = 512, C = 2, Kt = 200, EMB = 300;

  // outputs: z, recon_v, zx, zc, theta (flat, in return order)
  float* out    = (float*)d_out;
  float* z_out  = out;
  float* recon  = out + (size_t)N * C;
  float* zx_out = recon + (size_t)N * V;
  float* zc_out = zx_out + (size_t)N * C;
  float* theta  = zc_out + (size_t)Kt * C;

  // workspace layout (~93 MB)
  float* ws = (float*)d_ws;
  size_t off = 0;
  float* en1        = ws + off; off += (size_t)N * E1;
  float* en2        = ws + off; off += (size_t)N * E2;
  float* mean_lin   = ws + off; off += (size_t)N * C;
  float* logvar_lin = ws + off; off += (size_t)N * C;
  float* mu1        = ws + off; off += (size_t)Kt * 100;
  float* mu2        = ws + off; off += (size_t)Kt * 100;
  float* mu_z       = ws + off; off += (size_t)Kt * EMB;
  float* logits     = ws + off; off += (size_t)Kt * V;
  float* beta       = ws + off;

  dim3 blk(256);
  // en1 = softplus(input @ en1_W^T + b)  — the dominant 2.1e11-FLOP GEMM
  gemm_wmma_bf16<true, 1><<<dim3((E1 + 63) / 64, (N + 127) / 128), blk, 0, stream>>>(
      input_, en1_W, en1_b, en1, N, E1, V);
  // en2 = softplus(en1 @ en2_W^T + b)
  gemm_wmma_bf16<true, 1><<<dim3((E2 + 63) / 64, (N + 127) / 128), blk, 0, stream>>>(
      en1, en2_W, en2_b, en2, N, E2, E1);
  // mean/logvar heads (C=2)
  meanlogvar_kernel<<<N, 128, 0, stream>>>(en2, mean_W, mean_b, logvar_W, logvar_b,
                                           mean_lin, logvar_lin, E2);
  // fused BN + reparameterization + BN(z) + BN(topics)
  latent_kernel<<<1, 1024, 0, stream>>>(mean_lin, logvar_lin, eps, topics,
                                        g_mean, b_mean, g_logvar, b_logvar,
                                        g_x, b_x, g_phi, b_phi,
                                        z_out, zx_out, zc_out, N, Kt);
  // theta = softmax(-0.5 * ||zx - zc||^2)
  theta_kernel<<<N, 256, 0, stream>>>(zx_out, zc_out, theta, Kt);
  // mu chain (tiny)
  small_gemm_bt<1><<<(Kt * 100 + 255) / 256, 256, 0, stream>>>(zc_out, mu1_W, mu1_b, mu1, Kt, 100, C);
  small_gemm_bt<1><<<(Kt * 100 + 255) / 256, 256, 0, stream>>>(mu1, mu2_W, mu2_b, mu2, Kt, 100, 100);
  small_gemm_bt<0><<<(Kt * EMB + 255) / 256, 256, 0, stream>>>(mu2, mu_W, mu_b, mu_z, Kt, EMB, 100);
  // logits = mu_z @ emb^T   (emb is [V x EMB] = [N x K] form)
  gemm_wmma_bf16<true, 0><<<dim3((V + 63) / 64, (Kt + 127) / 128), blk, 0, stream>>>(
      mu_z, emb, nullptr, logits, Kt, V, EMB);
  // per-topic BN over V + softmax over V
  beta_kernel<<<Kt, 1024, 0, stream>>>(logits, beta_bias, g_dec, b_dec, beta, V);
  // recon_v = theta @ beta  (beta row-major [K x V]) — bandwidth-bound on 410 MB output
  gemm_wmma_bf16<false, 0><<<dim3((V + 63) / 64, (N + 127) / 128), blk, 0, stream>>>(
      theta, beta, nullptr, recon, N, V, Kt);
}
